// Transformer_45973329936525
// MI455X (gfx1250) — matrix operand order repaired and checked by reading, not compiled, not run on hardware
//
#include <hip/hip_runtime.h>
#include <stdint.h>

// ---------------- CDNA5 types ----------------
typedef __bf16 bf16;
typedef __attribute__((ext_vector_type(16))) __bf16 v16bf;
typedef __attribute__((ext_vector_type(8)))  __bf16 v8bf;
typedef __attribute__((ext_vector_type(4)))  __bf16 v4bf;
typedef __attribute__((ext_vector_type(8)))  float  v8f;

#define SOS_TOK 9998   // V - 2

__device__ inline bf16 f2bf(float x) {
  union { float f; unsigned u; } a; a.f = x;
  unsigned r = (a.u + 0x7FFFu + ((a.u >> 16) & 1u)) >> 16;  // RNE
  union { unsigned short s; bf16 b; } o; o.s = (unsigned short)r;
  return o.b;
}

__device__ inline v8f zero_v8f() {
  v8f z;
#pragma unroll
  for (int i = 0; i < 8; i++) z[i] = 0.f;
  return z;
}

__device__ inline v16bf cat16(v8bf lo, v8bf hi) {
  v16bf r;
#pragma unroll
  for (int i = 0; i < 8; i++) { r[i] = lo[i]; r[i + 8] = hi[i]; }
  return r;
}

// A-matrix fragment 16x32 bf16 from LDS [row][k] (row-major, ld halves).
// ISA 7.12.2: lanes 0-15 m=lane, K {0..7,16..23}; lanes 16-31 K {8..15,24..31}.
__device__ inline v16bf lda_frag(const bf16* base, int row0, int ld, int k0, int lane) {
  const bf16* p = base + (size_t)(row0 + (lane & 15)) * ld + k0 + ((lane >> 4) << 3);
  v8bf lo = *(const v8bf*)p;
  v8bf hi = *(const v8bf*)(p + 16);
  return cat16(lo, hi);
}

// B-matrix fragment 32x16 bf16 from LDS stored TRANSPOSED [n][k] (ld halves).
// Lanes 0-15: K {0..7,16..23}, lanes 16-31: K {8..15,24..31}, N = lane&15.
__device__ inline v16bf ldb_frag(const bf16* base, int n0, int ld, int k0, int lane) {
  const bf16* p = base + (size_t)(n0 + (lane & 15)) * ld + k0 + ((lane >> 4) << 3);
  v8bf lo = *(const v8bf*)p;
  v8bf hi = *(const v8bf*)(p + 16);
  return cat16(lo, hi);
}

__device__ inline v8f wmma_bf16(v16bf a, v16bf b, v8f c) {
  return __builtin_amdgcn_wmma_f32_16x16x32_bf16(false, a, false, b, (short)0, c, false, false);
}

// CDNA5 async global->LDS copy (16B per lane), tracked by ASYNCcnt (ISA 15.18.3 op 98).
__device__ inline void async_load_b128(unsigned lds_byte_off, const void* gaddr) {
  asm volatile("global_load_async_to_lds_b128 %0, %1, off"
               :: "v"(lds_byte_off), "v"((unsigned long long)(uintptr_t)gaddr)
               : "memory");
}
__device__ inline void wait_asynccnt0() {
  asm volatile("s_wait_asynccnt 0" ::: "memory");
}

// ---------------- weight pack: W f32 [K,N] -> Wt bf16 [Npad,K] (transposed, zero-padded rows) ----------------
__global__ void __launch_bounds__(256) pack_w_kernel(const float* __restrict__ W,
                                                     bf16* __restrict__ Wt,
                                                     int K, int N, int Npad) {
  size_t idx = (size_t)blockIdx.x * 256 + threadIdx.x;
  size_t total = (size_t)Npad * K;
  if (idx >= total) return;
  int n = (int)(idx / (size_t)K), k = (int)(idx % (size_t)K);
  Wt[idx] = (n < N) ? f2bf(W[(size_t)k * N + n]) : f2bf(0.f);
}

// ---------------- GEMM: C[M,N] = A[M,K] @ B[K,N] (+bias)(+relu) ----------------
// A: f32 row-major (converted in-flight). Bt: pre-packed bf16 [Npad][K].
// BM=BN=128, BK=64, 256 threads (8 waves), double-buffered LDS:
//   B tile: async global->LDS (byte-exact copy), A tile: float4 regs -> cvt -> ds_store.
__global__ void __launch_bounds__(256) gemm_kernel(
    const float* __restrict__ A, const bf16* __restrict__ Bt,
    const float* __restrict__ bias, float* __restrict__ C,
    int M, int N, int K, int relu) {
  __shared__ __align__(16) bf16 sA[2][128 * 64];
  __shared__ __align__(16) bf16 sB[2][128 * 64];   // [n][k]
  const int tid = threadIdx.x;
  const int lane = tid & 31, wave = tid >> 5;
  const int m0 = blockIdx.x * 128;
  const int n0 = blockIdx.y * 128;
  const int wm = (wave & 3) * 32;
  const int wn = (wave >> 2) * 64;
  const int NT = K >> 6;

  // B staging: thread -> row br = tid>>1, 64B half bh; 4 async b128 each.
  const int br = tid >> 1, bh = (tid & 1) * 32;                 // element offsets
  const bf16* bSrcBase = Bt + (size_t)(n0 + br) * K + bh;       // + kb later
  const unsigned bLdsRel = (unsigned)((br * 64 + bh) * 2);      // bytes within buffer

  float4 areg[8];

  auto issueB = [&](int kb, int buf) {
    const char* src = (const char*)(bSrcBase + kb);
    unsigned dst = (unsigned)(uintptr_t)(&sB[buf][0]) + bLdsRel;
#pragma unroll
    for (int j = 0; j < 4; j++)
      async_load_b128(dst + 16u * j, src + 16 * j);
  };
  auto loadA = [&](int kb) {
#pragma unroll
    for (int j = 0; j < 8; j++) {
      int idx4 = tid + 256 * j;
      int r = idx4 >> 4, c4 = idx4 & 15;
      int gm = m0 + r;
      areg[j] = (gm < M) ? *(const float4*)(A + (size_t)gm * K + kb + c4 * 4)
                         : make_float4(0.f, 0.f, 0.f, 0.f);
    }
  };
  auto storeA = [&](int buf) {
#pragma unroll
    for (int j = 0; j < 8; j++) {
      int idx4 = tid + 256 * j;
      int r = idx4 >> 4, c4 = idx4 & 15;
      v4bf b;
      b[0] = f2bf(areg[j].x); b[1] = f2bf(areg[j].y);
      b[2] = f2bf(areg[j].z); b[3] = f2bf(areg[j].w);
      *(v4bf*)(&sA[buf][r * 64 + c4 * 4]) = b;
    }
  };

  v8f acc[2][4];
#pragma unroll
  for (int i = 0; i < 2; i++)
#pragma unroll
    for (int j = 0; j < 4; j++) acc[i][j] = zero_v8f();

  // prolog: fill buffer 0
  issueB(0, 0);
  loadA(0);
  storeA(0);
  wait_asynccnt0();
  __syncthreads();

  for (int t = 0; t < NT; t++) {
    const int cur = t & 1, nxt = cur ^ 1;
    if (t + 1 < NT) {                 // overlap next-tile fetch with WMMA
      issueB((t + 1) << 6, nxt);
      loadA((t + 1) << 6);
    }
    const bf16* aB = &sA[cur][0];
    const bf16* bB = &sB[cur][0];
#pragma unroll
    for (int ks = 0; ks < 2; ks++) {
      v16bf af0 = lda_frag(aB, wm, 64, ks * 32, lane);
      v16bf af1 = lda_frag(aB, wm + 16, 64, ks * 32, lane);
#pragma unroll
      for (int nt = 0; nt < 4; nt++) {
        v16bf bf_ = ldb_frag(bB, wn + nt * 16, 64, ks * 32, lane);
        acc[0][nt] = wmma_bf16(af0, bf_, acc[0][nt]);
        acc[1][nt] = wmma_bf16(af1, bf_, acc[1][nt]);
      }
    }
    if (t + 1 < NT) storeA(nxt);
    wait_asynccnt0();
    __syncthreads();
  }

  // Epilogue. C/D layout: VGPR r -> M = r + 8*(lane>=16), N = lane&15.
  const int nl = lane & 15;
  const int rb = (lane >> 4) << 3;
#pragma unroll
  for (int mt = 0; mt < 2; mt++)
#pragma unroll
    for (int nt = 0; nt < 4; nt++) {
      int col = n0 + wn + nt * 16 + nl;
      if (col >= N) continue;
      float bv = bias ? bias[col] : 0.f;
#pragma unroll
      for (int r = 0; r < 8; r++) {
        int row = m0 + wm + mt * 16 + rb + r;
        if (row < M) {
          float v = acc[mt][nt][r] + bv;
          if (relu) v = fmaxf(v, 0.f);
          C[(size_t)row * N + col] = v;
        }
      }
    }
}

// ---------------- Causal MHA (N=32,H=8,T=256,DH=64), WMMA both GEMMs ----------------
__global__ void __launch_bounds__(256) attn_kernel(
    const float* __restrict__ Q, const float* __restrict__ K,
    const float* __restrict__ V, float* __restrict__ O) {
  __shared__ __align__(16) unsigned char smem[61440];
  float* S  = (float*)smem;                 // 32768 B
  bf16*  P  = (bf16*)(smem + 32768);        // 16384 B
  bf16*  Qs = (bf16*)(smem + 49152);        //  4096 B
  bf16*  Ks = (bf16*)(smem + 53248);        //  8192 B
  bf16*  Vt = (bf16*)smem;                  // aliases S in phase C

  const int qt = blockIdx.x, h = blockIdx.y, n = blockIdx.z;
  const int tid = threadIdx.x, lane = tid & 31, wave = tid >> 5;
  const int q0 = qt * 32;
  const float scale = 0.044194173824159216f;  // 1/sqrt(512)

  const float* qB = Q + (size_t)(n * 256) * 512 + h * 64;
  for (int idx = tid; idx < 32 * 64; idx += 256) {
    int r = idx >> 6, c = idx & 63;
    Qs[idx] = f2bf(qB[(size_t)(q0 + r) * 512 + c]);
  }

  const int mt = wave & 1, ntl = wave >> 1;
  for (int kt = 0; kt < 4; kt++) {
    __syncthreads();
    const float* kB = K + (size_t)(n * 256 + kt * 64) * 512 + h * 64;
    for (int idx = tid; idx < 64 * 64; idx += 256) {
      int r = idx >> 6, c = idx & 63;
      Ks[idx] = f2bf(kB[(size_t)r * 512 + c]);   // row-major K tile == B^T LDS layout
    }
    __syncthreads();
    v8f acc = zero_v8f();
#pragma unroll
    for (int ks = 0; ks < 2; ks++) {
      v16bf af = lda_frag(Qs, mt * 16, 64, ks * 32, lane);
      v16bf bf_ = ldb_frag(Ks, ntl * 16, 64, ks * 32, lane);
      acc = wmma_bf16(af, bf_, acc);
    }
    const int kcol = kt * 64 + ntl * 16 + (lane & 15);
    const int rbase = mt * 16 + ((lane >> 4) << 3);
#pragma unroll
    for (int r = 0; r < 8; r++) {
      int row = rbase + r;
      float e = acc[r] * scale;
      if (kcol > q0 + row) e = -1e20f;  // causal mask
      S[row * 256 + kcol] = e;
    }
  }
  __syncthreads();

  {  // row softmax: 8 lanes per row
    int row = tid >> 3, l8 = tid & 7;
    float* Sr = S + row * 256;
    float mx = -3.0e38f;
#pragma unroll
    for (int j = 0; j < 32; j++) mx = fmaxf(mx, Sr[l8 + j * 8]);
    mx = fmaxf(mx, __shfl_xor(mx, 1, 32));
    mx = fmaxf(mx, __shfl_xor(mx, 2, 32));
    mx = fmaxf(mx, __shfl_xor(mx, 4, 32));
    float sum = 0.f;
#pragma unroll
    for (int j = 0; j < 32; j++) {
      float e = __expf(Sr[l8 + j * 8] - mx);
      Sr[l8 + j * 8] = e;
      sum += e;
    }
    sum += __shfl_xor(sum, 1, 32);
    sum += __shfl_xor(sum, 2, 32);
    sum += __shfl_xor(sum, 4, 32);
    float inv = 1.f / sum;
    bf16* Pr = P + row * 256;
#pragma unroll
    for (int j = 0; j < 32; j++) Pr[l8 + j * 8] = f2bf(Sr[l8 + j * 8] * inv);
  }
  __syncthreads();

  const float* vB = V + (size_t)(n * 256) * 512 + h * 64;
  for (int idx = tid; idx < 64 * 256; idx += 256) {
    int kj = idx >> 6, d = idx & 63;
    Vt[d * 256 + kj] = f2bf(vB[(size_t)kj * 512 + d]);
  }
  __syncthreads();
  const int mt2 = wave & 1, nt2 = wave >> 1;
  v8f oacc = zero_v8f();
#pragma unroll
  for (int ks = 0; ks < 8; ks++) {
    v16bf af = lda_frag(P, mt2 * 16, 256, ks * 32, lane);
    v16bf bf_ = ldb_frag(Vt, nt2 * 16, 256, ks * 32, lane);
    oacc = wmma_bf16(af, bf_, oacc);
  }
  float* oB = O + (size_t)(n * 256) * 512 + h * 64;
  const int nl = lane & 15;
  const int rbase = mt2 * 16 + ((lane >> 4) << 3);
#pragma unroll
  for (int r = 0; r < 8; r++) {
    int row = q0 + rbase + r;
    oB[(size_t)row * 512 + nt2 * 16 + nl] = oacc[r];
  }
}

// ---------------- LayerNorm(a + b) * g + beta; bmode=1 => b broadcast per batch (row>>8) ----------------
__global__ void __launch_bounds__(256) ln_kernel(
    const float* __restrict__ a, const float* __restrict__ b,
    const float* __restrict__ g, const float* __restrict__ bt,
    float* __restrict__ out, int bmode) {
  const int row = blockIdx.x, tid = threadIdx.x;
  const float* ar = a + (size_t)row * 512;
  const float* br = b + (size_t)(bmode ? (row >> 8) : row) * 512;
  float x0 = ar[tid] + br[tid];
  float x1 = ar[tid + 256] + br[tid + 256];
  __shared__ float red[8];
  float s = x0 + x1;
#pragma unroll
  for (int o = 16; o >= 1; o >>= 1) s += __shfl_xor(s, o, 32);
  if ((tid & 31) == 0) red[tid >> 5] = s;
  __syncthreads();
  float mean = 0.f;
#pragma unroll
  for (int i = 0; i < 8; i++) mean += red[i];
  mean *= (1.f / 512.f);
  float d0 = x0 - mean, d1 = x1 - mean;
  float v = d0 * d0 + d1 * d1;
#pragma unroll
  for (int o = 16; o >= 1; o >>= 1) v += __shfl_xor(v, o, 32);
  __syncthreads();
  if ((tid & 31) == 0) red[tid >> 5] = v;
  __syncthreads();
  float var = 0.f;
#pragma unroll
  for (int i = 0; i < 8; i++) var += red[i];
  var *= (1.f / 512.f);
  float inv = rsqrtf(var + 1e-5f);
  out[(size_t)row * 512 + tid] = g[tid] * d0 * inv + bt[tid];
  out[(size_t)row * 512 + tid + 256] = g[tid + 256] * d1 * inv + bt[tid + 256];
}

// ---------------- small helpers ----------------
__global__ void add_row_kernel(const float* __restrict__ a, const float* __restrict__ rowv,
                               float* __restrict__ out) {
  int r = blockIdx.x, t = threadIdx.x;
  out[r * 512 + t] = a[r * 512 + t] + rowv[t];
  out[r * 512 + t + 256] = a[r * 512 + t + 256] + rowv[t + 256];
}

__global__ void embed_kernel(const int* __restrict__ trg, const float* __restrict__ wordW,
                             const float* __restrict__ posW, float* __restrict__ x) {
  int t = blockIdx.x, n = blockIdx.y, d = threadIdx.x;
  int tok = (t == 0) ? SOS_TOK : trg[n * 255 + (t - 1)];
  float* xp = x + (size_t)(n * 256 + t) * 512;
  xp[d] = wordW[(size_t)tok * 512 + d] + posW[(size_t)t * 512 + d];
  xp[d + 256] = wordW[(size_t)tok * 512 + d + 256] + posW[(size_t)t * 512 + d + 256];
}

// ---------------- host ----------------
static void gemm(const float* A, const bf16* Bt, const float* bias, float* C,
                 int M, int N, int K, int relu, hipStream_t s) {
  int Np = (N + 127) & ~127;
  dim3 g((M + 127) / 128, Np / 128);
  gemm_kernel<<<g, 256, 0, s>>>(A, Bt, bias, C, M, N, K, relu);
}

extern "C" void kernel_launch(void* const* d_in, const int* in_sizes, int n_in,
                              void* d_out, int out_size, void* d_ws, size_t ws_size,
                              hipStream_t stream) {
  (void)in_sizes; (void)n_in; (void)out_size; (void)ws_size;
  const float* input   = (const float*)d_in[0];
  const int*   trg     = (const int*)d_in[1];
  const float* fc_W    = (const float*)d_in[2];
  const float* fc_b    = (const float*)d_in[3];
  const float* enc_pos = (const float*)d_in[4];
  auto E = [&](int i, int j) { return (const float*)d_in[5 + 13 * i + j]; };
  const float* dec_word = (const float*)d_in[83];
  const float* dec_pos  = (const float*)d_in[84];
  auto Dd = [&](int i, int j) { return (const float*)d_in[85 + 20 * i + j]; };
  const float* out_W = (const float*)d_in[205];
  const float* out_b = (const float*)d_in[206];

  float* ws = (float*)d_ws;
  const size_t SZ = (size_t)8192 * 512;
  float* X    = ws;            // decoder activations [8192,512]
  float* Abuf = ws + SZ;
  float* Bbuf = ws + 2 * SZ;
  float* Cbuf = ws + 3 * SZ;
  float* Qb   = ws + 4 * SZ;
  float* Kb   = ws + 5 * SZ;
  float* Vb   = ws + 6 * SZ;
  float* AO   = ws + 7 * SZ;
  float* Hbuf = Qb;            // [8192,2048] reuses Q|K|V|AO after attention
  float* e0 = ws + 8 * SZ;     // encoder activations [32,512]
  float* e1 = e0 + 32 * 512;
  float* e2 = e1 + 32 * 512;
  float* eh = e2 + 32 * 512;   // [32,2048]

  // ---- packed bf16 weights (transposed [Npad][K]) after the f32 region ----
  bf16* wtb = (bf16*)(eh + 32 * 2048);
  size_t wcur = 0;
  auto pack = [&](const float* W, int K, int N) {
    int Np = (N + 127) & ~127;
    bf16* dst = wtb + wcur;
    wcur += (size_t)Np * K;
    size_t tot = (size_t)Np * K;
    pack_w_kernel<<<(unsigned)((tot + 255) / 256), 256, 0, stream>>>(W, dst, K, N, Np);
    return dst;
  };

  bf16* fcT = pack(fc_W, 2048, 512);
  bf16 *eWv[6], *eWo[6], *eW1[6], *eW2[6];
  for (int i = 0; i < 6; i++) {
    eWv[i] = pack(E(i, 2), 512, 512);
    eWo[i] = pack(E(i, 3), 512, 512);
    eW1[i] = pack(E(i, 9), 512, 2048);
    eW2[i] = pack(E(i, 11), 2048, 512);
  }
  bf16 *dWq[6], *dWk[6], *dWv[6], *dWo[6], *cWv[6], *cWo[6], *dW1[6], *dW2[6];
  for (int i = 0; i < 6; i++) {
    dWq[i] = pack(Dd(i, 0), 512, 512);
    dWk[i] = pack(Dd(i, 1), 512, 512);
    dWv[i] = pack(Dd(i, 2), 512, 512);
    dWo[i] = pack(Dd(i, 3), 512, 512);
    cWv[i] = pack(Dd(i, 9), 512, 512);
    cWo[i] = pack(Dd(i, 10), 512, 512);
    dW1[i] = pack(Dd(i, 16), 512, 2048);
    dW2[i] = pack(Dd(i, 18), 2048, 512);
  }
  bf16* outT = pack(out_W, 512, 10000);

  // ---- Encoder (seq len 1: self-attn == (x@Wv)@Wo + bo) ----
  gemm(input, fcT, fc_b, e1, 32, 512, 2048, 1, stream);
  add_row_kernel<<<32, 256, 0, stream>>>(e1, enc_pos, e0);
  for (int i = 0; i < 6; i++) {
    gemm(e0, eWv[i], nullptr, e1, 32, 512, 512, 0, stream);
    gemm(e1, eWo[i], E(i, 4), e2, 32, 512, 512, 0, stream);
    ln_kernel<<<32, 256, 0, stream>>>(e2, e0, E(i, 5), E(i, 6), e1, 0);
    gemm(e1, eW1[i], E(i, 10), eh, 32, 2048, 512, 1, stream);
    gemm(eh, eW2[i], E(i, 12), e2, 32, 512, 2048, 0, stream);
    ln_kernel<<<32, 256, 0, stream>>>(e2, e1, E(i, 7), E(i, 8), e0, 0);
  }

  // ---- Decoder ----
  embed_kernel<<<dim3(256, 32), 256, 0, stream>>>(trg, dec_word, dec_pos, X);
  for (int i = 0; i < 6; i++) {
    gemm(X, dWq[i], nullptr, Qb, 8192, 512, 512, 0, stream);
    gemm(X, dWk[i], nullptr, Kb, 8192, 512, 512, 0, stream);
    gemm(X, dWv[i], nullptr, Vb, 8192, 512, 512, 0, stream);
    attn_kernel<<<dim3(8, 8, 32), 256, 0, stream>>>(Qb, Kb, Vb, AO);
    gemm(AO, dWo[i], Dd(i, 4), Abuf, 8192, 512, 512, 0, stream);
    ln_kernel<<<8192, 256, 0, stream>>>(Abuf, X, Dd(i, 5), Dd(i, 6), Bbuf, 0);
    // cross-attention, kv_len==1 -> softmax==1 -> (enc@Wv)@Wo + bo, broadcast per batch
    gemm(e0, cWv[i], nullptr, e1, 32, 512, 512, 0, stream);
    gemm(e1, cWo[i], Dd(i, 11), e2, 32, 512, 512, 0, stream);
    ln_kernel<<<8192, 256, 0, stream>>>(Bbuf, e2, Dd(i, 12), Dd(i, 13), Cbuf, 1);
    // feed-forward
    gemm(Cbuf, dW1[i], Dd(i, 17), Hbuf, 8192, 2048, 512, 1, stream);
    gemm(Hbuf, dW2[i], Dd(i, 19), Abuf, 8192, 512, 2048, 0, stream);
    ln_kernel<<<8192, 256, 0, stream>>>(Abuf, Cbuf, Dd(i, 14), Dd(i, 15), X, 0);
  }

  // ---- Output projection ----
  gemm(X, outT, out_b, (float*)d_out, 8192, 10000, 512, 0, stream);
}
